// WindowAttention_9964324127345
// MI455X (gfx1250) — compile-verified, matrix operand order checked by reading, not run on hardware
//
#include <hip/hip_runtime.h>

typedef __attribute__((ext_vector_type(16))) __bf16 v16bf;
typedef __attribute__((ext_vector_type(8)))  float  v8f;

#define DEV static __device__ __forceinline__

// f32 -> bf16 (round-half-up) ; packed pair via v_perm_b32 when available.
#if __has_builtin(__builtin_amdgcn_perm)
DEV unsigned pk2(float a, float b) {
  unsigned ua = __builtin_bit_cast(unsigned, a) + 0x8000u;
  unsigned ub = __builtin_bit_cast(unsigned, b) + 0x8000u;
  return __builtin_amdgcn_perm(ub, ua, 0x07060302u);  // {b.hi16, a.hi16}
}
#else
DEV unsigned pk2(float a, float b) {
  unsigned ua = __builtin_bit_cast(unsigned, a) + 0x8000u;
  unsigned ub = __builtin_bit_cast(unsigned, b) + 0x8000u;
  return (ua >> 16) | (ub & 0xFFFF0000u);
}
#endif
DEV unsigned short f2bf(float x) {
  return (unsigned short)((__builtin_bit_cast(unsigned, x) + 0x8000u) >> 16);
}

DEV void sched_fence() {
#if __has_builtin(__builtin_amdgcn_sched_barrier)
  __builtin_amdgcn_sched_barrier(0);
#endif
}

DEV v8f wmma_bf16(v16bf a, v16bf b, v8f c) {
  // (neg_a, A, neg_b, B, c_mod, C, reuse_a, reuse_b)
  return __builtin_amdgcn_wmma_f32_16x16x32_bf16(false, a, false, b, (short)0, c, false, false);
}

DEV v8f zero8() {
  v8f z;
#pragma unroll
  for (int i = 0; i < 8; ++i) z[i] = 0.f;
  return z;
}

// 16x32 bf16 A/B fragment from a row-major bf16 matrix in LDS.
// Lane L holds row (L&15); K-halves {0..7,16..23} for L<16, {8..15,24..31} for L>=16.
DEV v16bf frag_lds(const unsigned short* m, int row0, int ld, int col0, int lane15, int hi) {
  const unsigned short* p = m + (row0 + lane15) * ld + col0 + hi * 8;
  union { v16bf v; uint4 u[2]; } f;
  f.u[0] = *(const uint4*)(p);
  f.u[1] = *(const uint4*)(p + 16);
  return f.v;
}

// Raw (pre-conversion) fragment source: 16 f32 from a row-major f32 global matrix.
struct rawf { float4 a, b, c, d; };
DEV rawf raw_gf32(const float* m, int row0, int ld, int col0, int lane15, int hi) {
  const float* p = m + (size_t)(row0 + lane15) * ld + col0 + hi * 8;
  rawf r;
  r.a = *(const float4*)(p);
  r.b = *(const float4*)(p + 4);
  r.c = *(const float4*)(p + 16);
  r.d = *(const float4*)(p + 20);
  return r;
}
DEV v16bf cvt_frag(const rawf& r) {
  union { v16bf v; unsigned u[8]; } f;
  f.u[0] = pk2(r.a.x, r.a.y); f.u[1] = pk2(r.a.z, r.a.w);
  f.u[2] = pk2(r.b.x, r.b.y); f.u[3] = pk2(r.b.z, r.b.w);
  f.u[4] = pk2(r.c.x, r.c.y); f.u[5] = pk2(r.c.z, r.c.w);
  f.u[6] = pk2(r.d.x, r.d.y); f.u[7] = pk2(r.d.z, r.d.w);
  return f.v;
}

#define NTOK 128
#define CDIM 192
#define NHEAD 6
#define HDIM 32

__global__ __launch_bounds__(256) void win_attn_fused(
    const float* __restrict__ x, const float* __restrict__ mask,
    const float* __restrict__ qkv_w, const float* __restrict__ proj_w,
    const float* __restrict__ proj_b, const float* __restrict__ rpb,
    const int* __restrict__ rel_index, float* __restrict__ out) {
  // LDS: q,k (row-major bf16), v^T (per-head [c][token]), attn-out, scratch(x then P)
  __shared__ unsigned short qs [NTOK * CDIM];   // 48 KB (pre-scaled by hd^-0.5)
  __shared__ unsigned short ks [NTOK * CDIM];   // 48 KB
  __shared__ unsigned short vts[CDIM * NTOK];   // 48 KB  vts[c*128 + m]
  __shared__ unsigned short aos[NTOK * CDIM];   // 48 KB  attention output (bf16)
  __shared__ unsigned short scr[NTOK * CDIM];   // 48 KB  phase1: x bf16; phase2: P strips (32 KB)

  const int tid = threadIdx.x;
  const int wave = __builtin_amdgcn_readfirstlane(tid >> 5);  // wave-uniform -> SGPR
  const int lane = tid & 31;
  const int lane15 = lane & 15;
  const int hi = lane >> 4;
  const int b = blockIdx.x;
  const float* xg = x + (size_t)b * (NTOK * CDIM);

  // Prefetch first qkv_w strip (global, independent of LDS) before the staging barrier.
  rawf rw[6];
#pragma unroll
  for (int kk = 0; kk < 6; ++kk)
    rw[kk] = raw_gf32(qkv_w, wave * 16, CDIM, kk * 32, lane15, hi);

  // ---- Phase 0: stage x tile as bf16 ----
  for (int i = tid; i < (NTOK * CDIM) / 4; i += 256) {
    float4 v = *(const float4*)(xg + i * 4);
    unsigned* d = (unsigned*)(scr + i * 4);
    d[0] = pk2(v.x, v.y);
    d[1] = pk2(v.z, v.w);
  }
  __syncthreads();

  // ---- Phase 1: QKV = x @ qkv_w^T (36 n-tiles; waves own n-strips, W read once/window) ----
  const float scale = 0.17677669529663687f;  // 32^-0.5, folded into q
  for (int nt = wave; nt < 36; nt += 8) {
    v16bf bw[6];
#pragma unroll
    for (int kk = 0; kk < 6; ++kk) bw[kk] = cvt_frag(rw[kk]);
    if (nt + 8 < 36) {  // prefetch next strip's W while this strip computes
#pragma unroll
      for (int kk = 0; kk < 6; ++kk)
        rw[kk] = raw_gf32(qkv_w, (nt + 8) * 16, CDIM, kk * 32, lane15, hi);
    }
    v16bf af0[6], af1[6];
#pragma unroll
    for (int kk = 0; kk < 6; ++kk)
      af0[kk] = frag_lds(scr, 0, CDIM, kk * 32, lane15, hi);
    sched_fence();
#pragma unroll
    for (int mt = 0; mt < 8; ++mt) {
      v16bf (&cur)[6] = (mt & 1) ? af1 : af0;
      v16bf (&nxt)[6] = (mt & 1) ? af0 : af1;
      if (mt < 7) {  // double-buffer: issue next A-frag loads above this WMMA chain
#pragma unroll
        for (int kk = 0; kk < 6; ++kk)
          nxt[kk] = frag_lds(scr, (mt + 1) * 16, CDIM, kk * 32, lane15, hi);
      }
      sched_fence();
      v8f acc = zero8();
#pragma unroll
      for (int kk = 0; kk < 6; ++kk)
        acc = wmma_bf16(cur[kk], bw[kk], acc);
      if (nt < 12) {            // q block: pre-scale (scalar branch: nt is SGPR)
#pragma unroll
        for (int j = 0; j < 8; ++j)
          qs[(mt * 16 + j + 8 * hi) * CDIM + nt * 16 + lane15] = f2bf(acc[j] * scale);
      } else if (nt < 24) {     // k block
#pragma unroll
        for (int j = 0; j < 8; ++j)
          ks[(mt * 16 + j + 8 * hi) * CDIM + (nt - 12) * 16 + lane15] = f2bf(acc[j]);
      } else {                  // v block: store transposed vts[c][m] (contiguous in j -> b128)
        const int c0 = (nt - 24) * 16 + lane15;
#pragma unroll
        for (int j = 0; j < 8; ++j)
          vts[c0 * NTOK + mt * 16 + j + 8 * hi] = f2bf(acc[j]);
      }
    }
  }
  __syncthreads();

  // ---- Phase 2: per-head attention; wave w owns token rows 16w..16w+15 ----
  const int wm = b & 63;  // mask window index
  const float* maskw = mask + (size_t)wm * NTOK * NTOK;
  unsigned short* P = scr;  // 128x128 bf16 exp-strip, per-wave rows

  // Hoist h-invariant gathers: rel-index table offsets and mask values, cached in registers.
  int   roff[8][8];
  float mval[8][8];
#pragma unroll
  for (int nt = 0; nt < 8; ++nt) {
#pragma unroll
    for (int j = 0; j < 8; ++j) {
      int M = wave * 16 + j + 8 * hi;
      int Nc = nt * 16 + lane15;
      roff[nt][j] = rel_index[M * NTOK + Nc] * NHEAD;
      mval[nt][j] = maskw[M * NTOK + Nc];
    }
  }

  for (int h = 0; h < NHEAD; ++h) {
    v16bf aq = frag_lds(qs, wave * 16, CDIM, h * HDIM, lane15, hi);
    // 1) C-init tiles: single gather per element (64 independent loads in flight)
    v8f s[8];
#pragma unroll
    for (int nt = 0; nt < 8; ++nt)
#pragma unroll
      for (int j = 0; j < 8; ++j)
        s[nt][j] = rpb[roff[nt][j] + h] + mval[nt][j];
    // 2) batch all k fragments, then S-WMMA chain (K == head dim == 32: one WMMA per tile)
    v16bf bk[8];
#pragma unroll
    for (int nt = 0; nt < 8; ++nt)
      bk[nt] = frag_lds(ks, nt * 16, CDIM, h * HDIM, lane15, hi);
    sched_fence();
#pragma unroll
    for (int nt = 0; nt < 8; ++nt)
      s[nt] = wmma_bf16(aq, bk[nt], s[nt]);
    // 3) row softmax: reduce across lanes of each 16-lane half (row M = j + 8*hi per VGPR j)
    float rs[8];
#pragma unroll
    for (int j = 0; j < 8; ++j) {
      float m = s[0][j];
#pragma unroll
      for (int nt = 1; nt < 8; ++nt) m = fmaxf(m, s[nt][j]);
#pragma unroll
      for (int d = 1; d < 16; d <<= 1) m = fmaxf(m, __shfl_xor(m, d, 32));
      float sum = 0.f;
#pragma unroll
      for (int nt = 0; nt < 8; ++nt) {
        float e = __expf(s[nt][j] - m);
        s[nt][j] = e;
        sum += e;
      }
#pragma unroll
      for (int d = 1; d < 16; d <<= 1) sum += __shfl_xor(sum, d, 32);
      rs[j] = 1.0f / sum;
    }
    // 4) store unnormalized exp strip as bf16 (A operand for E @ V)
#pragma unroll
    for (int nt = 0; nt < 8; ++nt)
#pragma unroll
      for (int j = 0; j < 8; ++j)
        P[(wave * 16 + j + 8 * hi) * NTOK + nt * 16 + lane15] = f2bf(s[nt][j]);
    asm volatile("" ::: "memory");  // keep DS stores before the fragment reloads

    // 5) O = E @ V_h : batch all fragments, then 8 WMMAs back-to-back
    v16bf ae[4], bv0[4], bv1[4];
#pragma unroll
    for (int kk = 0; kk < 4; ++kk) {
      ae[kk]  = frag_lds(P,   wave * 16,     NTOK, kk * 32, lane15, hi);
      bv0[kk] = frag_lds(vts, h * HDIM + 0,  NTOK, kk * 32, lane15, hi);
      bv1[kk] = frag_lds(vts, h * HDIM + 16, NTOK, kk * 32, lane15, hi);
    }
    sched_fence();
    v8f o0 = zero8(), o1 = zero8();
#pragma unroll
    for (int kk = 0; kk < 4; ++kk) {
      o0 = wmma_bf16(ae[kk], bv0[kk], o0);
      o1 = wmma_bf16(ae[kk], bv1[kk], o1);
    }
    // 6) normalize by row sum (same j <-> row mapping) and store bf16 attn-out
#pragma unroll
    for (int j = 0; j < 8; ++j) {
      int r = (wave * 16 + j + 8 * hi) * CDIM + h * HDIM;
      aos[r + lane15]      = f2bf(o0[j] * rs[j]);
      aos[r + 16 + lane15] = f2bf(o1[j] * rs[j]);
    }
  }
  __syncthreads();

  // ---- Phase 3: out = attn_out @ proj_w^T + proj_b ----
  float* og = out + (size_t)b * NTOK * CDIM;
  for (int nt = wave; nt < 12; nt += 8) {
    v16bf bw[6];
#pragma unroll
    for (int kk = 0; kk < 6; ++kk)
      bw[kk] = cvt_frag(raw_gf32(proj_w, nt * 16, CDIM, kk * 32, lane15, hi));
    float pb = proj_b[nt * 16 + lane15];
    v16bf af0[6], af1[6];
#pragma unroll
    for (int kk = 0; kk < 6; ++kk)
      af0[kk] = frag_lds(aos, 0, CDIM, kk * 32, lane15, hi);
    sched_fence();
#pragma unroll
    for (int mt = 0; mt < 8; ++mt) {
      v16bf (&cur)[6] = (mt & 1) ? af1 : af0;
      v16bf (&nxt)[6] = (mt & 1) ? af0 : af1;
      if (mt < 7) {
#pragma unroll
        for (int kk = 0; kk < 6; ++kk)
          nxt[kk] = frag_lds(aos, (mt + 1) * 16, CDIM, kk * 32, lane15, hi);
      }
      sched_fence();
      v8f acc;
#pragma unroll
      for (int j = 0; j < 8; ++j) acc[j] = pb;  // bias folded into C
#pragma unroll
      for (int kk = 0; kk < 6; ++kk)
        acc = wmma_bf16(cur[kk], bw[kk], acc);
#pragma unroll
      for (int j = 0; j < 8; ++j)
        og[(mt * 16 + j + 8 * hi) * CDIM + nt * 16 + lane15] = acc[j];
    }
  }
}

extern "C" void kernel_launch(void* const* d_in, const int* in_sizes, int n_in,
                              void* d_out, int out_size, void* d_ws, size_t ws_size,
                              hipStream_t stream) {
  const float* x      = (const float*)d_in[0];
  const float* mask   = (const float*)d_in[1];
  const float* qkv_w  = (const float*)d_in[2];
  const float* proj_w = (const float*)d_in[3];
  const float* proj_b = (const float*)d_in[4];
  const float* rpb    = (const float*)d_in[5];
  const int*   rel    = (const int*)d_in[6];
  float* out = (float*)d_out;
  win_attn_fused<<<1024, 256, 0, stream>>>(x, mask, qkv_w, proj_w, proj_b, rpb, rel, out);
}
